// Quantizer_23124103922124
// MI455X (gfx1250) — compile-verified, matrix operand order checked by reading, not run on hardware
//
#include <hip/hip_runtime.h>
#include <hip/hip_bf16.h>

// ---------------------------------------------------------------------------
// VQ-VAE quantizer, fused dist+argmin on CDNA5 WMMA (bf16 in / fp32 acc),
// with async global->LDS double-buffered codebook staging (ASYNCcnt path).
//   dist(n,k) = ||x_n||^2 + ||c_k||^2 - 2 x_n . c_k ; argmin_k drops ||x||^2.
//   quantized = codebook[argmin] (fp32 gather), loss = 1.25 * mean((x-q)^2).
// ---------------------------------------------------------------------------

typedef __bf16 bf16_t;
typedef __attribute__((ext_vector_type(16))) __bf16 v16bf;
typedef __attribute__((ext_vector_type(8)))  __bf16 v8bf;
typedef __attribute__((ext_vector_type(8)))  float  v8f;
typedef __attribute__((ext_vector_type(4)))  int    v4i;

#define D_DIM      64
#define CHUNK      128   // codebook rows per LDS buffer (double-buffered)
#define ROW_STRIDE 72    // bf16 elems per LDS row (64 + 8 pad) -> 144B, 16B aligned, conflict-free

#define AS1 __attribute__((address_space(1)))
#define AS3 __attribute__((address_space(3)))

// --- async global->LDS helpers (CDNA5 ASYNCcnt path) -----------------------
// builtin sig (from clang diagnostics): (T __device__* src, T __shared__* dst, Imm off, Imm cpol)
__device__ __forceinline__ void async_copy_b128(const void* gsrc, void* ldst) {
#if __has_builtin(__builtin_amdgcn_global_load_async_to_lds_b128)
  __builtin_amdgcn_global_load_async_to_lds_b128(
      (AS1 v4i*)gsrc, (AS3 v4i*)ldst, 0, 0);
#else
  unsigned loff = (unsigned)(unsigned long long)(AS3 void*)ldst;
  asm volatile("global_load_async_to_lds_b128 %0, %1, off"
               :: "v"(loff), "v"(gsrc) : "memory");
#endif
}

__device__ __forceinline__ void async_copy_b32(const void* gsrc, void* ldst) {
#if __has_builtin(__builtin_amdgcn_global_load_async_to_lds_b32)
  __builtin_amdgcn_global_load_async_to_lds_b32(
      (AS1 int*)gsrc, (AS3 int*)ldst, 0, 0);
#else
  unsigned loff = (unsigned)(unsigned long long)(AS3 void*)ldst;
  asm volatile("global_load_async_to_lds_b32 %0, %1, off"
               :: "v"(loff), "v"(gsrc) : "memory");
#endif
}

__device__ __forceinline__ void wait_async0() {
#if __has_builtin(__builtin_amdgcn_s_wait_asynccnt)
  __builtin_amdgcn_s_wait_asynccnt(0);
#else
  asm volatile("s_wait_asynccnt 0x0" ::: "memory");
#endif
}

// --- prep: codebook fp32 -> bf16 (row major) + ||c_k||^2 -------------------
__global__ __launch_bounds__(32) void prep_cb_kernel(const float* __restrict__ cb,
                                                     bf16_t* __restrict__ cbb,
                                                     float* __restrict__ cnorm) {
  const int code = blockIdx.x;
  const int lane = threadIdx.x;              // 32 lanes, 2 elems each
  const float2 v = ((const float2*)(cb + (size_t)code * D_DIM))[lane];
  cbb[(size_t)code * D_DIM + lane * 2 + 0] = (bf16_t)v.x;
  cbb[(size_t)code * D_DIM + lane * 2 + 1] = (bf16_t)v.y;
  float s = v.x * v.x + v.y * v.y;
  #pragma unroll
  for (int m = 1; m < 32; m <<= 1) s += __shfl_xor(s, m, 32);
  if (lane == 0) cnorm[code] = s;
}

// A-matrix lane fill (16x32 bf16, ISA 7.12.2): element j holds K = kb+kbase+j+8*(j>=8)
__device__ __forceinline__ v16bf load_a_tile(const float* __restrict__ xrow, int kb, int kbase) {
  const float* p0 = xrow + kb + kbase;       // K runs [base..base+7]
  const float* p1 = p0 + 16;                 // and [base+16..base+23]
  v16bf a;
  #pragma unroll
  for (int j = 0; j < 8; ++j) { a[j] = (bf16_t)p0[j]; a[j + 8] = (bf16_t)p1[j]; }
  return a;
}

// B-matrix lane fill from LDS (column = codebook row, same K packing as A)
__device__ __forceinline__ v16bf load_b_tile(const bf16_t* p) {
  v8bf lo = *(const v8bf*)(p);               // ds_read_b128
  v8bf hi = *(const v8bf*)(p + 16);          // ds_read_b128
  v16bf b;
  #pragma unroll
  for (int j = 0; j < 8; ++j) { b[j] = lo[j]; b[j + 8] = hi[j]; }
  return b;
}

__global__ __launch_bounds__(256) void vq_argmin_kernel(
    const float* __restrict__ x, const float* __restrict__ cbf32,
    const bf16_t* __restrict__ cbb, const float* __restrict__ cnorm,
    float* __restrict__ out_q, float* __restrict__ out_idx,
    float* __restrict__ partial, int K) {
  __shared__ bf16_t lds_cb[2][CHUNK * ROW_STRIDE];   // 2 x 18KB, padded rows
  __shared__ float  lds_cn[2][CHUNK];
  __shared__ float  wsum[8];

  const int tid   = threadIdx.x;
  const int wave  = tid >> 5;
  const int lane  = tid & 31;
  const int r     = lane & 15;               // column / row-in-tile index
  const int half  = lane >> 4;               // 0: K base 0, 1: K base 8
  const int kbase = half * 8;

  const int rowT0 = blockIdx.x * 256 + wave * 32;   // this wave: rows [rowT0, rowT0+32)
  const int rowT1 = rowT0 + 16;

  // A tiles: 2 row-tiles x 2 K-chunks (D=64), loaded once, reused for all of K.
  const v16bf a00 = load_a_tile(x + (size_t)(rowT0 + r) * D_DIM, 0,  kbase);
  const v16bf a01 = load_a_tile(x + (size_t)(rowT0 + r) * D_DIM, 32, kbase);
  const v16bf a10 = load_a_tile(x + (size_t)(rowT1 + r) * D_DIM, 0,  kbase);
  const v16bf a11 = load_a_tile(x + (size_t)(rowT1 + r) * D_DIM, 32, kbase);

  float best0[8], best1[8];
  int   idx0[8],  idx1[8];
  #pragma unroll
  for (int i = 0; i < 8; ++i) {
    best0[i] = __builtin_huge_valf(); best1[i] = __builtin_huge_valf();
    idx0[i] = 0; idx1[i] = 0;
  }

  // stage chunk `cbase` into buffer `buf` via per-lane async copies
  auto stage = [&](int cbase, int buf) {
    #pragma unroll
    for (int s = 0; s < (CHUNK * 8) / 256; ++s) {    // 4 x b128 per thread
      const int i = tid + s * 256;
      const int row = i >> 3, piece = i & 7;
      async_copy_b128(cbb + (size_t)(cbase + row) * D_DIM + piece * 8,
                      lds_cb[buf] + row * ROW_STRIDE + piece * 8);
    }
    if (tid < CHUNK)
      async_copy_b32(cnorm + cbase + tid, &lds_cn[buf][tid]);
  };

  const int nchunks = K / CHUNK;
  stage(0, 0);                                       // prologue

  for (int ci = 0; ci < nchunks; ++ci) {
    const int buf = ci & 1;
    wait_async0();           // this wave's async writes for chunk ci landed
    __syncthreads();         // all waves: data visible, prior buffer readers done
    if (ci + 1 < nchunks) stage((ci + 1) * CHUNK, buf ^ 1);

    const int cbase = ci * CHUNK;
    for (int t = 0; t < CHUNK / 16; ++t) {
      const bf16_t* rowp = lds_cb[buf] + (t * 16 + r) * ROW_STRIDE + kbase;
      const v16bf b0 = load_b_tile(rowp);        // K 0..31
      const v16bf b1 = load_b_tile(rowp + 32);   // K 32..63
      v8f acc0 = {0.f,0.f,0.f,0.f,0.f,0.f,0.f,0.f};
      v8f acc1 = {0.f,0.f,0.f,0.f,0.f,0.f,0.f,0.f};
      acc0 = __builtin_amdgcn_wmma_f32_16x16x32_bf16(false, a00, false, b0, (short)0, acc0, false, false);
      acc0 = __builtin_amdgcn_wmma_f32_16x16x32_bf16(false, a01, false, b1, (short)0, acc0, false, false);
      acc1 = __builtin_amdgcn_wmma_f32_16x16x32_bf16(false, a10, false, b0, (short)0, acc1, false, false);
      acc1 = __builtin_amdgcn_wmma_f32_16x16x32_bf16(false, a11, false, b1, (short)0, acc1, false, false);

      const float cn   = lds_cn[buf][t * 16 + r]; // column norm, same for all 8 acc slots
      const int   code = cbase + t * 16 + r;
      #pragma unroll
      for (int i = 0; i < 8; ++i) {
        const float d0 = fmaf(-2.f, acc0[i], cn);
        if (d0 < best0[i]) { best0[i] = d0; idx0[i] = code; }   // strict < keeps first occurrence
        const float d1 = fmaf(-2.f, acc1[i], cn);
        if (d1 < best1[i]) { best1[i] = d1; idx1[i] = code; }
      }
    }
  }

  // cross-lane argmin (16 lanes share a row), then gather + loss
  float lsum = 0.f;
  #pragma unroll
  for (int rt = 0; rt < 2; ++rt) {
    const int rowBase = rt ? rowT1 : rowT0;
    #pragma unroll
    for (int i = 0; i < 8; ++i) {
      float v = rt ? best1[i] : best0[i];
      int  id = rt ? idx1[i]  : idx0[i];
      #pragma unroll
      for (int m = 1; m < 16; m <<= 1) {       // butterfly within each 16-lane half
        const float ov = __shfl_xor(v, m, 32);
        const int  oid = __shfl_xor(id, m, 32);
        if (ov < v || (ov == v && oid < id)) { v = ov; id = oid; }
      }
      const int row = rowBase + i + half * 8;  // C layout: VGPR i = rows i / i+8
      if (r == 0) out_idx[row] = (float)id;
      const float4 qv = ((const float4*)(cbf32 + (size_t)id * D_DIM))[r];
      const float4 xv = ((const float4*)(x + (size_t)row * D_DIM))[r];
      ((float4*)(out_q + (size_t)row * D_DIM))[r] = qv;
      const float e0 = xv.x - qv.x, e1 = xv.y - qv.y, e2 = xv.z - qv.z, e3 = xv.w - qv.w;
      lsum += e0 * e0 + e1 * e1 + e2 * e2 + e3 * e3;
    }
  }
  #pragma unroll
  for (int m = 1; m < 32; m <<= 1) lsum += __shfl_xor(lsum, m, 32);
  if (lane == 0) wsum[wave] = lsum;
  __syncthreads();
  if (tid == 0) {
    float s = 0.f;
    #pragma unroll
    for (int i = 0; i < 8; ++i) s += wsum[i];
    partial[blockIdx.x] = s;
  }
}

// deterministic final loss reduction (fixed-order tree; no float atomics)
__global__ __launch_bounds__(256) void vq_loss_reduce(const float* __restrict__ partial,
                                                      int n, float* __restrict__ loss_out,
                                                      float scale) {
  __shared__ float s[256];
  float v = 0.f;
  for (int i = threadIdx.x; i < n; i += 256) v += partial[i];
  s[threadIdx.x] = v;
  __syncthreads();
  for (int st = 128; st > 0; st >>= 1) {
    if (threadIdx.x < st) s[threadIdx.x] += s[threadIdx.x + st];
    __syncthreads();
  }
  if (threadIdx.x == 0) loss_out[0] = s[0] * scale;
}

extern "C" void kernel_launch(void* const* d_in, const int* in_sizes, int n_in,
                              void* d_out, int out_size, void* d_ws, size_t ws_size,
                              hipStream_t stream) {
  const float* x  = (const float*)d_in[0];
  const float* cb = (const float*)d_in[1];
  const int N = in_sizes[0] / D_DIM;   // 65536
  const int K = in_sizes[1] / D_DIM;   // 4096

  float* out      = (float*)d_out;
  float* out_q    = out;                                // N*64
  float* out_idx  = out + (size_t)N * D_DIM;            // N
  float* out_loss = out + (size_t)N * D_DIM + N;        // 1

  // workspace carve: bf16 codebook (K*64*2B) | cnorm (K*4B) | partials (N/256*4B)
  char*   ws      = (char*)d_ws;
  bf16_t* cbb     = (bf16_t*)ws;
  float*  cnorm   = (float*)(ws + (size_t)K * D_DIM * sizeof(bf16_t));
  float*  partial = cnorm + K;

  prep_cb_kernel<<<K, 32, 0, stream>>>(cb, cbb, cnorm);
  const int nblocks = N / 256;
  vq_argmin_kernel<<<nblocks, 256, 0, stream>>>(x, cb, cbb, cnorm, out_q, out_idx, partial, K);
  vq_loss_reduce<<<1, 256, 0, stream>>>(partial, nblocks, out_loss,
                                        1.25f / ((float)N * (float)D_DIM));
}
